// NTMAdresser_20246475834105
// MI455X (gfx1250) — matrix operand order repaired and checked by reading, not compiled, not run on hardware
//
#include <hip/hip_runtime.h>
#include <hip/hip_bf16.h>
#include <math.h>

// ---------------------------------------------------------------------------
// Types (plain clang ext vectors; trivially-copyable so unions are legal)
// ---------------------------------------------------------------------------
typedef __attribute__((ext_vector_type(16))) __bf16    v16bf;
typedef __attribute__((ext_vector_type(8)))  float     v8f;
typedef __attribute__((ext_vector_type(4)))  float     f32x4;
typedef __attribute__((ext_vector_type(4)))  unsigned  u32x4;
typedef __attribute__((ext_vector_type(2)))  unsigned  u32x2;

#define B_DIM 128
#define E_DIM 512
#define H_DIM 1024
#define S_DIM (E_DIM + 2)
#define NEG_MASK -1000000.0f
#define MROWS 32          // key rows per block (2 WMMA m-tiles share each B frag)
#define ABF_STRIDE 1032   // bf16 A-tile row stride (1024 + 8 -> conflict-free b128)

__device__ __forceinline__ unsigned bf16pack(float a, float b) {
  unsigned ua = __float_as_uint(a);
  unsigned ub = __float_as_uint(b);
  ua = (ua + 0x7FFFu + ((ua >> 16) & 1u)) >> 16;  // RNE
  ub = (ub + 0x7FFFu + ((ub >> 16) & 1u)) >> 16;
  return (ub << 16) | (ua & 0xFFFFu);
}

__device__ __forceinline__ float gelu_exact(float x) {
  return 0.5f * x * (1.0f + erff(x * 0.7071067811865476f));
}

// ---------------------------------------------------------------------------
// Kernel 1: pack W1_bot (rows H..2H-1 of W1) to bf16 in WMMA-B tile order.
// Tile (kt, nt): element (kk, nn) at lane*16+s, lane = nn + 16*((kk>>3)&1),
// s = (kk&7) + 8*(kk>>4) -> lane reads contiguous 32B (column N=lane%16, table K order).
// ---------------------------------------------------------------------------
__global__ __launch_bounds__(256) void pack_w1_kernel(
    const float* __restrict__ W1, unsigned short* __restrict__ w1bf) {
  int i = blockIdx.x * 256 + threadIdx.x;  // 0 .. 1048575
  int k = i >> 10;
  int n = i & 1023;
  float f = W1[(size_t)(H_DIM + k) * H_DIM + n];
  unsigned u = __float_as_uint(f);
  u = (u + 0x7FFFu + ((u >> 16) & 1u)) >> 16;
  int kt = k >> 5, kk = k & 31, nt = n >> 4, nn = n & 15;
  w1bf[(((size_t)kt * 64 + nt) << 9) + (nn + (((kk >> 3) & 1) << 4)) * 16 + (kk & 7) + ((kk >> 4) << 3)] = (unsigned short)u;
}

// ---------------------------------------------------------------------------
// Kernel 2: qW[b][n] = q[b] . W1_top[:,n] + b1[n];  nullW/newW = addr . W1_bot
// ---------------------------------------------------------------------------
__global__ __launch_bounds__(256) void qproj_kernel(
    const float* __restrict__ query, const float* __restrict__ W1,
    const float* __restrict__ b1, const float* __restrict__ nullA,
    const float* __restrict__ newA, float* __restrict__ qWb,
    float* __restrict__ nullW, float* __restrict__ newW) {
  const int rowId = blockIdx.x;          // 0..127 = batch, 128 = null, 129 = new
  const int n0 = threadIdx.x << 2;
  const float* x;
  const float* w;
  if (rowId < B_DIM)        { x = query + (size_t)rowId * H_DIM; w = W1; }
  else if (rowId == B_DIM)  { x = nullA; w = W1 + (size_t)H_DIM * H_DIM; }
  else                      { x = newA;  w = W1 + (size_t)H_DIM * H_DIM; }
  float a0 = 0.f, a1 = 0.f, a2 = 0.f, a3 = 0.f;
  for (int d = 0; d < H_DIM; ++d) {
    float xv = x[d];
    const float* wr = w + (size_t)d * H_DIM + n0;
    a0 = __builtin_fmaf(xv, wr[0], a0);
    a1 = __builtin_fmaf(xv, wr[1], a1);
    a2 = __builtin_fmaf(xv, wr[2], a2);
    a3 = __builtin_fmaf(xv, wr[3], a3);
  }
  if (rowId < B_DIM) {
    float* o = qWb + (size_t)rowId * H_DIM + n0;
    o[0] = a0 + b1[n0]; o[1] = a1 + b1[n0 + 1];
    o[2] = a2 + b1[n0 + 2]; o[3] = a3 + b1[n0 + 3];
  } else {
    float* o = (rowId == B_DIM ? nullW : newW) + n0;
    o[0] = a0; o[1] = a1; o[2] = a2; o[3] = a3;
  }
}

// ---------------------------------------------------------------------------
// Kernel 3 (hot): fused  raw[b,e] = sum_n gelu((keys@W1bot)[m,n]+qW[b,n])*W2[n]
// One block per 32-row keys tile. Keys async-copied (ASYNCcnt) to LDS in two
// 16-row halves, converted ONCE to a bf16 A-tile in LDS; k-loop is pure
// ds_load_b128 (A) + global b128 (B) + v_wmma. Both m-tiles share each B frag.
// ---------------------------------------------------------------------------
union BFrag { u32x4 q[2]; v16bf v; };

__global__ __launch_bounds__(256) void gemm_score_kernel(
    const float* __restrict__ keys, const unsigned short* __restrict__ w1bf,
    const float* __restrict__ qWb, const float* __restrict__ W2,
    float* __restrict__ raw) {
  __shared__ float stage[16 * 1024];                 // 64 KB fp32 staging
  __shared__ unsigned short abf[MROWS * ABF_STRIDE]; // 64.5 KB bf16 A tile
  __shared__ float scoreLds[MROWS];

  const int tid  = threadIdx.x;
  const int wave = tid >> 5;
  const int lane = tid & 31;
  const int b    = blockIdx.x >> 4;          // 16 tiles of 32 rows per batch
  const int e0   = (blockIdx.x & 15) << 5;

  if (tid < MROWS) scoreLds[tid] = 0.0f;

  // ---- stage + convert the 32x1024 fp32 keys tile, 16 rows at a time ----
  for (int h = 0; h < 2; ++h) {
    const float* gbase = keys + ((size_t)b * E_DIM + e0 + h * 16) * H_DIM;
    for (int it = 0; it < 16; ++it) {
      int idx = it * 256 + tid;              // 4096 chunks of 16B
      int row = idx >> 8;                    // 256 chunks per row
      int c   = (idx & 255) << 2;            // float column
      unsigned ldsOff = (unsigned)(size_t)&stage[row * 1024 + c];
      const float* g  = gbase + row * H_DIM + c;
      asm volatile("global_load_async_to_lds_b128 %0, %1, off"
                   :: "v"(ldsOff), "v"(g) : "memory");
    }
    asm volatile("s_wait_asynccnt 0x0" ::: "memory");
    __syncthreads();
    // convert fp32 -> bf16 once (shared by all 8 waves afterwards)
    for (int it = 0; it < 16; ++it) {
      int idx = it * 256 + tid;
      int row = idx >> 8;
      int c   = (idx & 255) << 2;
      f32x4 x = *(const f32x4*)&stage[row * 1024 + c];
      u32x2 w;
      w[0] = bf16pack(x[0], x[1]);
      w[1] = bf16pack(x[2], x[3]);
      *(u32x2*)&abf[(h * 16 + row) * ABF_STRIDE + c] = w;
    }
    __syncthreads();
  }

  // ---- WMMA main loop: wave owns 8 N-tiles x 2 M-tiles ----
  const int wnt0  = wave << 3;
  const int arow  = lane & 15;               // A-matrix row M within m-tile
  const int khalf = (lane >> 4) << 3;        // 0 (lanes 0-15) or 8 (16-31)

  v8f acc[16] = {};                          // acc[2*t + mt]
  for (int kt = 0; kt < 32; ++kt) {
    // A fragments: contiguous 16B ds loads per ISA 16-bit 16x32 layout
    BFrag a0, a1;
    const unsigned short* a0p = &abf[arow * ABF_STRIDE + (kt << 5) + khalf];
    a0.q[0] = *(const u32x4*)a0p;
    a0.q[1] = *(const u32x4*)(a0p + 16);
    const unsigned short* a1p = a0p + 16 * ABF_STRIDE;
    a1.q[0] = *(const u32x4*)a1p;
    a1.q[1] = *(const u32x4*)(a1p + 16);

    const u32x4* bp =
        (const u32x4*)(w1bf + (((size_t)(kt << 6) + wnt0) << 9)) + (lane << 1);
#pragma unroll
    for (int t = 0; t < 8; ++t) {
      BFrag bf;
      bf.q[0] = bp[t * 64];
      bf.q[1] = bp[t * 64 + 1];
      acc[2 * t] = __builtin_amdgcn_wmma_f32_16x16x32_bf16(
          false, a0.v, false, bf.v, (short)0, acc[2 * t], false, false);
      acc[2 * t + 1] = __builtin_amdgcn_wmma_f32_16x16x32_bf16(
          false, a1.v, false, bf.v, (short)0, acc[2 * t + 1], false, false);
    }
  }

  // ---- fused epilogue: +qW, exact gelu, dot with W2, reduce over n ----
  const float* qrow = qWb + (size_t)b * H_DIM;
  float partial[16] = {};  // m = mt*16 + khalf + v
#pragma unroll
  for (int t = 0; t < 8; ++t) {
    int n = ((wnt0 + t) << 4) + arow;
    float qw = qrow[n];
    float w2 = W2[n];
#pragma unroll
    for (int mt = 0; mt < 2; ++mt) {
#pragma unroll
      for (int v = 0; v < 8; ++v) {
        float x = acc[2 * t + mt][v] + qw;
        partial[mt * 8 + v] = __builtin_fmaf(gelu_exact(x), w2,
                                             partial[mt * 8 + v]);
      }
    }
  }
#pragma unroll
  for (int off = 8; off >= 1; off >>= 1)
#pragma unroll
    for (int i = 0; i < 16; ++i)
      partial[i] += __shfl_xor(partial[i], off, 32);

  if (arow == 0) {  // lanes 0 and 16 hold per-half sums
#pragma unroll
    for (int v = 0; v < 8; ++v) {
      atomicAdd(&scoreLds[khalf + v], partial[v]);
      atomicAdd(&scoreLds[16 + khalf + v], partial[8 + v]);
    }
  }
  __syncthreads();
  if (tid < MROWS) raw[(size_t)b * E_DIM + e0 + tid] = scoreLds[tid];
}

// ---------------------------------------------------------------------------
// Kernel 4: null (slot 0) and new (slot E+1) scores. grid = B*2 blocks.
// ---------------------------------------------------------------------------
__global__ __launch_bounds__(256) void nullnew_kernel(
    const float* __restrict__ nullW, const float* __restrict__ newW,
    const float* __restrict__ qWb, const float* __restrict__ W2,
    const float* __restrict__ b2, const float* __restrict__ mention,
    float* __restrict__ out) {
  const int b = blockIdx.x >> 1;
  const int slot = blockIdx.x & 1;  // 0 = null, 1 = new
  const float* aw = slot ? newW : nullW;
  __shared__ float red[256];
  float s = 0.0f;
  for (int n = threadIdx.x; n < H_DIM; n += 256) {
    float x = aw[n] + qWb[(size_t)b * H_DIM + n];
    s = __builtin_fmaf(gelu_exact(x), W2[n], s);
  }
  red[threadIdx.x] = s;
  __syncthreads();
  for (int st = 128; st > 0; st >>= 1) {
    if (threadIdx.x < st) red[threadIdx.x] += red[threadIdx.x + st];
    __syncthreads();
  }
  if (threadIdx.x == 0) {
    float sc = red[0] + b2[0] + (slot ? mention[b] : -mention[b]);
    out[(size_t)b * S_DIM + (slot ? (S_DIM - 1) : 0)] = sc;
  }
}

// ---------------------------------------------------------------------------
// Kernel 5: key slots s=1..E: +b2, +mention, entity-size mask.
// ---------------------------------------------------------------------------
__global__ __launch_bounds__(256) void finalize_kernel(
    const float* __restrict__ raw, const int* __restrict__ entsz,
    const float* __restrict__ mention, const float* __restrict__ b2,
    float* __restrict__ out) {
  int i = blockIdx.x * 256 + threadIdx.x;  // 65536
  int b = i >> 9;
  int e = i & (E_DIM - 1);
  float v = raw[i] + b2[0] + mention[b];
  if (e >= entsz[b]) v += NEG_MASK;  // slot s=e+1 >= entity_size+1 (s<=E always)
  out[(size_t)b * S_DIM + 1 + e] = v;
}

// ---------------------------------------------------------------------------
extern "C" void kernel_launch(void* const* d_in, const int* in_sizes, int n_in,
                              void* d_out, int out_size, void* d_ws,
                              size_t ws_size, hipStream_t stream) {
  const float* query   = (const float*)d_in[0];
  const float* keys    = (const float*)d_in[1];
  const int*   entsz   = (const int*)d_in[2];
  const float* mention = (const float*)d_in[3];
  const float* W1      = (const float*)d_in[4];
  const float* b1      = (const float*)d_in[5];
  const float* W2      = (const float*)d_in[6];
  const float* b2      = (const float*)d_in[7];
  const float* nullA   = (const float*)d_in[8];
  const float* newA    = (const float*)d_in[9];
  float* out = (float*)d_out;

  char* ws = (char*)d_ws;
  unsigned short* w1bf = (unsigned short*)ws;                // 2 MB bf16 tiles
  float* qWb   = (float*)(ws + (size_t)(2u << 20));          // 512 KB
  float* nullW = (float*)(ws + (size_t)(2u << 20) + (512u << 10));
  float* newW  = nullW + H_DIM;                              // 4 KB each
  float* raw   = newW + H_DIM;                               // 256 KB

  pack_w1_kernel<<<(H_DIM * H_DIM) / 256, 256, 0, stream>>>(W1, w1bf);
  qproj_kernel<<<B_DIM + 2, 256, 0, stream>>>(query, W1, b1, nullA, newA,
                                              qWb, nullW, newW);
  gemm_score_kernel<<<B_DIM * (E_DIM / MROWS), 256, 0, stream>>>(keys, w1bf,
                                                                 qWb, W2, raw);
  nullnew_kernel<<<B_DIM * 2, 256, 0, stream>>>(nullW, newW, qWb, W2, b2,
                                                mention, out);
  finalize_kernel<<<(B_DIM * E_DIM) / 256, 256, 0, stream>>>(raw, entsz,
                                                             mention, b2, out);
}
